// RWKV_Tmix_x070_54760833024019
// MI455X (gfx1250) — matrix operand order repaired and checked by reading, not compiled, not run on hardware
//
#include <hip/hip_runtime.h>
#include <math.h>

#define Bsz 4
#define Tn  1024
#define Cn  1024
#define Hn  16
#define Nn  64
#define Mrows (Bsz * Tn)

typedef __attribute__((ext_vector_type(16))) __bf16        v16bf;
typedef __attribute__((ext_vector_type(8)))  float         v8f;
typedef __attribute__((ext_vector_type(8)))  unsigned int  v8u;

// round-to-nearest-even f32 -> bf16 (stored as u16)
static __device__ __forceinline__ unsigned short f2bf(float f) {
  union { float f; unsigned u; } v; v.f = f;
  unsigned r = v.u + 0x7FFFu + ((v.u >> 16) & 1u);
  return (unsigned short)(r >> 16);
}

// LDS byte offset of a shared-memory pointer (flat shared addresses map to the
// low 32 bits per the aperture rules; compile-target semantics only).
static __device__ __forceinline__ unsigned ldsoff(const void* p) {
  return (unsigned)(uintptr_t)p;
}

// ---------------------------------------------------------------------------
__global__ void cast_bf16_kernel(const float* __restrict__ in,
                                 unsigned short* __restrict__ out, int n) {
  int i = blockIdx.x * blockDim.x + threadIdx.x;
  if (i < n) out[i] = f2bf(in[i]);
}

// transpose (R x Cc) f32 -> (Cc x R) bf16
__global__ void transpose_cast_kernel(const float* __restrict__ in,
                                      unsigned short* __restrict__ out,
                                      int R, int Cc) {
  int i = blockIdx.x * blockDim.x + threadIdx.x;
  if (i < R * Cc) {
    int r = i / Cc, c = i - r * Cc;
    out[c * R + r] = f2bf(in[i]);
  }
}

// ---------------------------------------------------------------------------
// Fused token-mix: one block per row m = b*T + t.
__global__ void __launch_bounds__(128) mix_kernel(
    const float* __restrict__ x,
    const float* __restrict__ maa_x,  const float* __restrict__ maa_rg,
    const float* __restrict__ maa_wa, const float* __restrict__ maa_k,
    const float* __restrict__ maa_v,
    const float* __restrict__ w1,     const float* __restrict__ w2,
    unsigned short* __restrict__ xrg, unsigned short* __restrict__ xwa,
    unsigned short* __restrict__ xk,  unsigned short* __restrict__ xv) {
  __shared__ float xs[Cn], xps[Cn], hs[Cn], mixin[128];
  const int m = blockIdx.x;
  const int t = m & (Tn - 1);
  const int tid = threadIdx.x;
  const size_t rowOff = (size_t)m * Cn;

#pragma unroll
  for (int i = 0; i < 8; ++i) {
    int c = tid + i * 128;
    float xc = x[rowOff + c];
    float xp = (t == 0) ? 0.0f : x[rowOff - Cn + c];
    xs[c] = xc; xps[c] = xp;
    hs[c] = xc + (xp - xc) * maa_x[c];
  }
  __syncthreads();
  {
    float s = 0.0f;
    for (int c = 0; c < Cn; ++c) s += hs[c] * w1[(size_t)c * 128 + tid];
    mixin[tid] = tanhf(s);
  }
  __syncthreads();
#pragma unroll
  for (int i = 0; i < 8; ++i) {
    int c = tid + i * 128;
    float xx = xps[c] - xs[c];
    float m0 = 0.f, m1 = 0.f, m2 = 0.f, m3 = 0.f;
#pragma unroll 8
    for (int d = 0; d < 32; ++d) {
      m0 += mixin[d]      * w2[(size_t)(d)      * Cn + c];
      m1 += mixin[32 + d] * w2[(size_t)(32 + d) * Cn + c];
      m2 += mixin[64 + d] * w2[(size_t)(64 + d) * Cn + c];
      m3 += mixin[96 + d] * w2[(size_t)(96 + d) * Cn + c];
    }
    float xb = xs[c];
    xrg[rowOff + c] = f2bf(xb + xx * (maa_rg[c] + m0));
    xwa[rowOff + c] = f2bf(xb + xx * (maa_wa[c] + m1));
    xk [rowOff + c] = f2bf(xb + xx * (maa_k[c]  + m2));
    xv [rowOff + c] = f2bf(xb + xx * (maa_v[c]  + m3));
  }
}

// ---------------------------------------------------------------------------
// bf16 WMMA GEMM: D(MxN) = A(MxK) * Bt(NxK)^T, f32 accumulate.
// Block tile 128x128, 8 waves; wave w computes 16 rows x 128 cols (8 WMMA accs).
// Full tiles (K%32==0, whole 128-col tile in range) are staged with the CDNA5
// async global->LDS engine (ASYNCcnt + s_wait_asynccnt), double buffered, one
// barrier per K-chunk. Ragged LoRA tiles use guarded synchronous staging with
// zero padding. LDS row stride = 20 dwords: 16B-aligned b128 writes and
// conflict-free 16-row fragment gathers (20*r mod 64 distinct for r=0..15).
#define LSTR 20

static __device__ __forceinline__ void wmma_chunk(
    const unsigned int* __restrict__ As, const unsigned int* __restrict__ Bs,
    v8f (&acc)[8], int wave, int half, int l16) {
  int arow = wave * 16 + l16;
  v8u au;
#pragma unroll
  for (int j = 0; j < 4; ++j) {
    au[j]     = As[arow * LSTR + half * 4 + j];
    au[j + 4] = As[arow * LSTR + 8 + half * 4 + j];
  }
  v16bf af = __builtin_bit_cast(v16bf, au);
#pragma unroll
  for (int f = 0; f < 8; ++f) {
    int bn = f * 16 + l16;
    v8u bu;
#pragma unroll
    for (int j = 0; j < 8; ++j) bu[j] = Bs[bn * LSTR + (j >> 2) * 8 + half * 4 + (j & 3)];
    v16bf bfrag = __builtin_bit_cast(v16bf, bu);
    acc[f] = __builtin_amdgcn_wmma_f32_16x16x32_bf16(
        false, af, false, bfrag, (short)0, acc[f], false, false);
  }
}

__global__ void __launch_bounds__(256) wmma_gemm_kernel(
    const unsigned short* __restrict__ A, const unsigned short* __restrict__ Bt,
    int M, int N, int K,
    float* __restrict__ outF, unsigned short* __restrict__ outB,
    const float* __restrict__ bias, const float* __restrict__ addend, int op) {
  __shared__ alignas(16) unsigned int As[2][128 * LSTR];
  __shared__ alignas(16) unsigned int Bs[2][128 * LSTR];
  const int tid = threadIdx.x;
  const int mBase = blockIdx.y * 128;
  const int nBase = blockIdx.x * 128;
  const int Kp = K >> 1;  // bf16 pairs per row
  const unsigned int* A32 = reinterpret_cast<const unsigned int*>(A);
  const unsigned int* B32 = reinterpret_cast<const unsigned int*>(Bt);
  const int lane = tid & 31, wave = tid >> 5;
  const int half = lane >> 4, l16 = lane & 15;
  const int nK = (K + 31) >> 5;

  v8f acc[8];
#pragma unroll
  for (int f = 0; f < 8; ++f) acc[f] = (v8f){};

  const bool fullTile = ((K & 31) == 0) && (nBase + 128 <= N);
  if (fullTile) {
    // -------- async double-buffered path --------
    // per chunk: A 128x16 dwords + B 128x16 dwords, staged as b128 async loads
    auto issue = [&](int kc, int buf) {
#pragma unroll
      for (int i = 0; i < 2; ++i) {  // A tile: 512 x b128 slots
        int slot = tid + i * 256;
        int row = slot >> 2, grp = slot & 3;
        unsigned long long ga = (unsigned long long)(uintptr_t)A32 +
            ((size_t)(mBase + row) * Kp + (size_t)kc * 16 + grp * 4) * 4ull;
        unsigned lo = ldsoff(&As[buf][row * LSTR + grp * 4]);
        asm volatile("global_load_async_to_lds_b128 %0, %1, off"
                     :: "v"(lo), "v"(ga) : "memory");
      }
#pragma unroll
      for (int i = 0; i < 2; ++i) {  // B tile
        int slot = tid + i * 256;
        int row = slot >> 2, grp = slot & 3;
        unsigned long long ga = (unsigned long long)(uintptr_t)B32 +
            ((size_t)(nBase + row) * Kp + (size_t)kc * 16 + grp * 4) * 4ull;
        unsigned lo = ldsoff(&Bs[buf][row * LSTR + grp * 4]);
        asm volatile("global_load_async_to_lds_b128 %0, %1, off"
                     :: "v"(lo), "v"(ga) : "memory");
      }
    };
    issue(0, 0);
    asm volatile("s_wait_asynccnt 0x0" ::: "memory");
    __syncthreads();
    for (int kc = 0; kc < nK; ++kc) {
      int cur = kc & 1;
      if (kc + 1 < nK) issue(kc + 1, cur ^ 1);
      wmma_chunk(As[cur], Bs[cur], acc, wave, half, l16);
      asm volatile("s_wait_asynccnt 0x0" ::: "memory");
      __syncthreads();
    }
  } else {
    // -------- guarded synchronous path (ragged LoRA tiles) --------
    for (int kc = 0; kc < nK; ++kc) {
#pragma unroll
      for (int i = 0; i < 8; ++i) {
        int slot = tid + i * 256;
        int row = slot >> 4, kp = slot & 15;
        int gk = kc * 16 + kp;
        unsigned int av = 0u, bv = 0u;
        if (gk < Kp) av = A32[(size_t)(mBase + row) * Kp + gk];
        int gn = nBase + row;
        if (gn < N && gk < Kp) bv = B32[(size_t)gn * Kp + gk];
        As[0][row * LSTR + kp] = av;
        Bs[0][row * LSTR + kp] = bv;
      }
      if (kc + 1 < nK) {
        int gk = (kc + 1) * 16 + (tid & 15);
        int prow = mBase + ((tid >> 4) << 3);
        if (gk < Kp) __builtin_prefetch(&A32[(size_t)prow * Kp + gk], 0, 0);
      }
      __syncthreads();
      wmma_chunk(As[0], Bs[0], acc, wave, half, l16);
      __syncthreads();
    }
  }

  // epilogue: D layout -> m = mBase + wave*16 + half*8 + v, n = nBase + f*16 + l16
#pragma unroll
  for (int f = 0; f < 8; ++f) {
    int n = nBase + f * 16 + l16;
    if (n < N) {
#pragma unroll
      for (int v = 0; v < 8; ++v) {
        int m = mBase + wave * 16 + half * 8 + v;
        float val = acc[f][v];
        if (addend) val += addend[(size_t)m * N + n];
        if (bias) val += bias[n];
        if (op == 1) {
          val = tanhf(val);
        } else if (op == 2) {
          val = 1.0f / (1.0f + __expf(-val));
        } else if (op == 3) {  // -softplus(-val) - 0.5, numerically stable
          val = fminf(val, 0.0f) - log1pf(__expf(-fabsf(val))) - 0.5f;
        }
        if (outF) outF[(size_t)m * N + n] = val;
        if (outB) outB[(size_t)m * N + n] = f2bf(val);
      }
    }
  }
}

// ---------------------------------------------------------------------------
// Fused wkv7 scan + kk-norm + k-mix + GroupNorm + bonus + gate.
__global__ void __launch_bounds__(64) wkv_gn_kernel(
    const float* __restrict__ Rb,  const float* __restrict__ Wdb,
    const float* __restrict__ K0b, const float* __restrict__ Vb,
    const float* __restrict__ KKb, const float* __restrict__ Ab,
    const float* __restrict__ MAb, const float* __restrict__ MK2b,
    const float* __restrict__ Gb,  const float* __restrict__ faaaa,
    const float* __restrict__ lnw, const float* __restrict__ lnb,
    unsigned short* __restrict__ ygb) {
  const int bh = blockIdx.x;
  const int b = bh >> 4, h = bh & 15;
  const int i = threadIdx.x;
  const int c = h * Nn + i;
  const float fa = faaaa[c];
  const float lw = lnw[c], lb = lnb[c];
  __shared__ float r_s[Nn], d_s[Nn], k_s[Nn], v_s[Nn], a_s[Nn], b_s[Nn];
  __shared__ float redA[Nn], redB[Nn], redC[Nn];
  float S[Nn];
#pragma unroll
  for (int j = 0; j < Nn; ++j) S[j] = 0.0f;

  size_t idx = ((size_t)b * Tn) * Cn + c;
  for (int t = 0; t < Tn; ++t, idx += Cn) {
    float rv  = Rb[idx],  wl  = Wdb[idx], k0 = K0b[idx], vv = Vb[idx];
    float kkp = KKb[idx], aa  = Ab[idx],  mav = MAb[idx], mk2 = MK2b[idx];
    float gv  = Gb[idx];

    redA[i] = kkp * kkp;
    __syncthreads();
    if (i < 32) {
      float ssum = redA[i] + redA[i + 32];
      for (int off = 16; off > 0; off >>= 1) ssum += __shfl_down(ssum, off, 32);
      if (i == 0) redA[0] = ssum;
    }
    __syncthreads();
    float inv = 1.0f / fmaxf(sqrtf(redA[0]), 1e-12f);
    float kkn = kkp * inv;
    float kf = (k0 * mav + k0 * aa * (1.0f - mav)) *
               __expf(fminf(wl * mk2, 0.0f));

    r_s[i] = rv; d_s[i] = __expf(wl); k_s[i] = kf; v_s[i] = vv;
    a_s[i] = -kkn; b_s[i] = kkn * aa;
    __syncthreads();

    float sa = 0.0f;
#pragma unroll
    for (int j = 0; j < Nn; ++j) sa += S[j] * a_s[j];
    float y = 0.0f;
#pragma unroll
    for (int j = 0; j < Nn; ++j) {
      float sv = S[j] * d_s[j] + sa * b_s[j] + vv * k_s[j];
      S[j] = sv;
      y += sv * r_s[j];
    }

    redA[i] = y; redB[i] = y * y; redC[i] = rv * kf * fa;
    __syncthreads();
    if (i < 32) {
      float s1 = redA[i] + redA[i + 32];
      float s2 = redB[i] + redB[i + 32];
      float s3 = redC[i] + redC[i + 32];
      for (int off = 16; off > 0; off >>= 1) {
        s1 += __shfl_down(s1, off, 32);
        s2 += __shfl_down(s2, off, 32);
        s3 += __shfl_down(s3, off, 32);
      }
      if (i == 0) { redA[0] = s1; redB[0] = s2; redC[0] = s3; }
    }
    __syncthreads();
    float mu  = redA[0] * (1.0f / 64.0f);
    float var = redB[0] * (1.0f / 64.0f) - mu * mu;
    float yn  = (y - mu) * rsqrtf(var + 6.4e-4f) * lw + lb;
    float yfin = (yn + redC[0] * vv) * gv;
    ygb[idx] = f2bf(yfin);
    __syncthreads();
  }
}

// ---------------------------------------------------------------------------
static inline void gemm(hipStream_t s, const unsigned short* A,
                        const unsigned short* Bt, int M, int N, int K,
                        float* outF, unsigned short* outB, const float* bias,
                        const float* addend, int op) {
  dim3 g((N + 127) / 128, M / 128);
  wmma_gemm_kernel<<<g, 256, 0, s>>>(A, Bt, M, N, K, outF, outB, bias, addend, op);
}

extern "C" void kernel_launch(void* const* d_in, const int* in_sizes, int n_in,
                              void* d_out, int out_size, void* d_ws, size_t ws_size,
                              hipStream_t stream) {
  (void)in_sizes; (void)n_in; (void)out_size; (void)ws_size;
  const float* x      = (const float*)d_in[0];
  const float* maa_x  = (const float*)d_in[1];
  const float* maa_rg = (const float*)d_in[2];
  const float* maa_wa = (const float*)d_in[3];
  const float* maa_k  = (const float*)d_in[4];
  const float* maa_v  = (const float*)d_in[5];
  const float* maa_w1 = (const float*)d_in[6];
  const float* maa_w2 = (const float*)d_in[7];
  const float* tdecay = (const float*)d_in[8];
  const float* dw1    = (const float*)d_in[9];
  const float* dw2    = (const float*)d_in[10];
  const float* faaaa  = (const float*)d_in[11];
  const float* aaaaa  = (const float*)d_in[12];
  const float* aw1    = (const float*)d_in[13];
  const float* aw2    = (const float*)d_in[14];
  const float* kw1    = (const float*)d_in[15];
  const float* kw2    = (const float*)d_in[16];
  const float* gw1    = (const float*)d_in[17];
  const float* gw2    = (const float*)d_in[18];
  const float* mawi1  = (const float*)d_in[19];
  const float* mawi2  = (const float*)d_in[20];
  const float* misc_a = (const float*)d_in[21];
  const float* mkw1   = (const float*)d_in[22];
  const float* mkw2   = (const float*)d_in[23];
  const float* misc_k = (const float*)d_in[24];
  const float* Wr     = (const float*)d_in[25];
  const float* Wk     = (const float*)d_in[26];
  const float* Wv     = (const float*)d_in[27];
  const float* Wo     = (const float*)d_in[28];
  const float* lnw    = (const float*)d_in[29];
  const float* lnb    = (const float*)d_in[30];

  char* base = (char*)d_ws;
  size_t off = 0;
  auto alloc = [&](size_t bytes) -> void* {
    void* p = base + off;
    off = (off + bytes + 255) & ~(size_t)255;
    return p;
  };
  const int M = Mrows, C = Cn;
  const size_t MCb = (size_t)M * C * sizeof(unsigned short);
  const size_t MCf = (size_t)M * C * sizeof(float);

  unsigned short* xrg_b = (unsigned short*)alloc(MCb);
  unsigned short* xwa_b = (unsigned short*)alloc(MCb);
  unsigned short* xk_b  = (unsigned short*)alloc(MCb);
  unsigned short* xv_b  = (unsigned short*)alloc(MCb);
  unsigned short* Wr_b  = (unsigned short*)alloc((size_t)C * C * 2);
  unsigned short* Wk_b  = (unsigned short*)alloc((size_t)C * C * 2);
  unsigned short* Wv_b  = (unsigned short*)alloc((size_t)C * C * 2);
  unsigned short* Wo_b  = (unsigned short*)alloc((size_t)C * C * 2);
  unsigned short* gw1t  = (unsigned short*)alloc((size_t)128 * C * 2);
  unsigned short* gw2t  = (unsigned short*)alloc((size_t)C * 128 * 2);
  unsigned short* dw1t  = (unsigned short*)alloc((size_t)64 * C * 2);
  unsigned short* dw2t  = (unsigned short*)alloc((size_t)C * 64 * 2);
  unsigned short* aw1t  = (unsigned short*)alloc((size_t)16 * C * 2);
  unsigned short* aw2t  = (unsigned short*)alloc((size_t)C * 16 * 2);
  unsigned short* maw1t = (unsigned short*)alloc((size_t)16 * C * 2);
  unsigned short* maw2t = (unsigned short*)alloc((size_t)C * 16 * 2);
  unsigned short* kw1t  = (unsigned short*)alloc((size_t)16 * C * 2);
  unsigned short* kw2t  = (unsigned short*)alloc((size_t)C * 16 * 2);
  unsigned short* mkw1t = (unsigned short*)alloc((size_t)16 * C * 2);
  unsigned short* mkw2t = (unsigned short*)alloc((size_t)C * 16 * 2);
  float* Rb     = (float*)alloc(MCf);
  float* K0b    = (float*)alloc(MCf);
  float* Vb     = (float*)alloc(MCf);
  float* Wdb    = (float*)alloc(MCf);
  float* Abuf   = (float*)alloc(MCf);
  float* MAbuf  = (float*)alloc(MCf);
  float* MK2buf = (float*)alloc(MCf);
  float* KKbuf  = (float*)alloc(MCf);
  float* Gbuf   = (float*)alloc(MCf);
  unsigned short* G1b  = (unsigned short*)alloc((size_t)M * 128 * 2);
  unsigned short* D1b  = (unsigned short*)alloc((size_t)M * 64 * 2);
  unsigned short* A1b  = (unsigned short*)alloc((size_t)M * 16 * 2);
  unsigned short* MA1b = (unsigned short*)alloc((size_t)M * 16 * 2);
  unsigned short* MK1b = (unsigned short*)alloc((size_t)M * 16 * 2);
  unsigned short* T1b  = (unsigned short*)alloc((size_t)M * 16 * 2);
  unsigned short* ygb  = (unsigned short*)alloc(MCb);

  // --- weight prep ---
  int nCC = C * C;
  cast_bf16_kernel<<<(nCC + 255) / 256, 256, 0, stream>>>(Wr, Wr_b, nCC);
  cast_bf16_kernel<<<(nCC + 255) / 256, 256, 0, stream>>>(Wk, Wk_b, nCC);
  cast_bf16_kernel<<<(nCC + 255) / 256, 256, 0, stream>>>(Wv, Wv_b, nCC);
  cast_bf16_kernel<<<(nCC + 255) / 256, 256, 0, stream>>>(Wo, Wo_b, nCC);
  auto tc = [&](const float* in, unsigned short* out, int R, int Cc) {
    int n = R * Cc;
    transpose_cast_kernel<<<(n + 255) / 256, 256, 0, stream>>>(in, out, R, Cc);
  };
  tc(gw1, gw1t, C, 128);  tc(gw2, gw2t, 128, C);
  tc(dw1, dw1t, C, 64);   tc(dw2, dw2t, 64, C);
  tc(aw1, aw1t, C, 16);   tc(aw2, aw2t, 16, C);
  tc(mawi1, maw1t, C, 16); tc(mawi2, maw2t, 16, C);
  tc(kw1, kw1t, C, 16);   tc(kw2, kw2t, 16, C);
  tc(mkw1, mkw1t, C, 16); tc(mkw2, mkw2t, 16, C);

  // --- token mixing ---
  mix_kernel<<<M, 128, 0, stream>>>(x, maa_x, maa_rg, maa_wa, maa_k, maa_v,
                                    maa_w1, maa_w2, xrg_b, xwa_b, xk_b, xv_b);

  // --- projections & LoRA chains (all WMMA) ---
  gemm(stream, xrg_b, Wr_b, M, C, C, Rb, nullptr, nullptr, nullptr, 0);
  gemm(stream, xk_b,  Wk_b, M, C, C, K0b, nullptr, nullptr, nullptr, 0);
  gemm(stream, xv_b,  Wv_b, M, C, C, Vb, nullptr, nullptr, nullptr, 0);
  gemm(stream, xrg_b, gw1t, M, 128, C, nullptr, G1b, nullptr, nullptr, 1);
  gemm(stream, G1b, gw2t, M, C, 128, Gbuf, nullptr, nullptr, nullptr, 0);
  gemm(stream, xwa_b, dw1t, M, 64, C, nullptr, D1b, nullptr, nullptr, 1);
  gemm(stream, D1b, dw2t, M, C, 64, Wdb, nullptr, tdecay, nullptr, 3);
  gemm(stream, xwa_b, aw1t, M, 16, C, nullptr, A1b, nullptr, nullptr, 0);
  gemm(stream, A1b, aw2t, M, C, 16, Abuf, nullptr, aaaaa, nullptr, 2);
  gemm(stream, xwa_b, maw1t, M, 16, C, nullptr, MA1b, nullptr, nullptr, 0);
  gemm(stream, MA1b, maw2t, M, C, 16, MAbuf, nullptr, misc_a, nullptr, 2);
  gemm(stream, xk_b, mkw1t, M, 16, C, nullptr, MK1b, nullptr, nullptr, 0);
  gemm(stream, MK1b, mkw2t, M, C, 16, MK2buf, nullptr, misc_k, nullptr, 2);
  gemm(stream, xk_b, kw1t, M, 16, C, nullptr, T1b, nullptr, nullptr, 1);
  gemm(stream, T1b, kw2t, M, C, 16, KKbuf, nullptr, nullptr, K0b, 0);

  // --- fused wkv scan + GroupNorm + bonus + gate ---
  wkv_gn_kernel<<<Bsz * Hn, Nn, 0, stream>>>(Rb, Wdb, K0b, Vb, KKbuf, Abuf,
                                             MAbuf, MK2buf, Gbuf, faaaa, lnw,
                                             lnb, ygb);

  // --- output projection into d_out (f32) ---
  gemm(stream, ygb, Wo_b, M, C, C, (float*)d_out, nullptr, nullptr, nullptr, 0);
}
